// RelativePositionalEncoding_53730040873024
// MI455X (gfx1250) — matrix operand order repaired and checked by reading, not compiled, not run on hardware
//
#include <hip/hip_runtime.h>
#include <hip/hip_fp16.h>
#include <stdint.h>

typedef __attribute__((ext_vector_type(16))) _Float16 v16h;
typedef __attribute__((ext_vector_type(8)))  float    v8f;
typedef __attribute__((ext_vector_type(4)))  float    v4f;   // native vec for NT stores

#define BB      8
#define TT      4096
#define NTOK    (BB * TT)       // 32768 tokens
#define DMODEL  1024
#define DHALF   512
#define DQ      256
#define MAXOBS  2000
#define GROWS   4112            // gaps 0..4095, padded to 257*16 for WMMA tiling
#define MTILES  (GROWS / 16)    // 257
#define NTILES  (DHALF / 16)    // 32
#define KSTEPS  (DQ / 32)       // 8

__device__ __forceinline__ float gelu_exact(float x) {
  return 0.5f * x * (1.0f + erff(x * 0.7071067811865475f));
}

// ---------------------------------------------------------------------------
// Kernel 1: per-batch-row scan. One block per row, 256 threads * 16 elements.
// Produces obs_count (clipped cumsum of valid - 1) and gap (t - prev valid t).
// ---------------------------------------------------------------------------
__global__ void scan_kernel(const uint8_t* __restrict__ mask,
                            int* __restrict__ obs, int* __restrict__ gap) {
  const int b    = blockIdx.x;
  const int tid  = threadIdx.x;        // 0..255
  const int CH   = TT / 256;           // 16
  const int base = tid * CH;
  const uint8_t* m = mask + (size_t)b * TT;

  uint8_t loc[CH];
  int cnt = 0, last = -1;
#pragma unroll
  for (int i = 0; i < CH; ++i) {
    loc[i] = m[base + i];              // nonzero == padded
    if (loc[i] == 0) { cnt++; last = base + i; }
  }

  __shared__ int scnt[256];
  __shared__ int slast[256];
  scnt[tid]  = cnt;
  slast[tid] = last;
  __syncthreads();
  // Hillis-Steele inclusive scan: (+) for counts, (max) for last-valid-index.
  for (int off = 1; off < 256; off <<= 1) {
    int c = scnt[tid], l = slast[tid];
    int c2 = 0, l2 = -1;
    if (tid >= off) { c2 = scnt[tid - off]; l2 = slast[tid - off]; }
    __syncthreads();
    scnt[tid]  = c + c2;
    slast[tid] = (l > l2) ? l : l2;
    __syncthreads();
  }
  int run_cnt  = (tid > 0) ? scnt[tid - 1]  : 0;    // exclusive prefixes
  int run_last = (tid > 0) ? slast[tid - 1] : -1;

#pragma unroll
  for (int i = 0; i < CH; ++i) {
    const int  t = base + i;
    const bool v = (loc[i] == 0);
    if (v) run_cnt++;
    int oc = run_cnt - 1;
    oc = oc < 0 ? 0 : (oc > (MAXOBS - 1) ? (MAXOBS - 1) : oc);
    const int gp = (v && run_last >= 0) ? (t - run_last) : 0;
    if (v) run_last = t;
    obs[(size_t)b * TT + t] = oc;
    gap[(size_t)b * TT + t] = gp;
  }
}

// ---------------------------------------------------------------------------
// WMMA lane layouts (ISA 7.12.2), pre-applied at pack time so the GEMM inner
// loop is pure 32B-per-lane contiguous loads:
//   A (16-bit, 16x32/k-step): lane = M%16 + 16*hi,
//       K%32 = r: r<8 -> (hi=0,e=r); r<16 -> (hi=1,e=r-8);
//               r<24 -> (hi=0,e=r-8); else (hi=1,e=r-16)
//   B (16-bit, 32x16/k-step): lane = N%16 + 16*hi, same K%32 -> (hi,e) map as A
// Packed index: chunk = (tile*KSTEPS + kk)*32 + lane, halves e = 0..15.
// ---------------------------------------------------------------------------
__global__ void prep_kernel(const float* __restrict__ w1, const float* __restrict__ b1,
                            const float* __restrict__ w2,
                            _Float16* __restrict__ Ap, _Float16* __restrict__ Bp) {
  const int i = blockIdx.x * blockDim.x + threadIdx.x;
  if (i < GROWS * DQ) {
    const int g = i >> 8;              // row (gap value), DQ == 256
    const int k = i & (DQ - 1);
    const float x = (float)g * w1[k] + b1[k];
    const _Float16 h = (_Float16)gelu_exact(x);
    const int mlo = g & 15, mt = g >> 4;
    const int kk = k >> 5, r = k & 31;
    int hi, e;
    if (r < 8)       { hi = 0; e = r; }
    else if (r < 16) { hi = 1; e = r - 8; }
    else if (r < 24) { hi = 0; e = r - 8; }
    else             { hi = 1; e = r - 16; }
    const int lane = mlo + 16 * hi;
    Ap[(((size_t)mt * KSTEPS + kk) * 32 + lane) * 16 + e] = h;
  }
  if (i < DQ * DHALF) {
    const int k = i >> 9;              // DHALF == 512
    const int n = i & (DHALF - 1);
    const int nlo = n & 15, nt = n >> 4;
    const int kk = k >> 5, r = k & 31;
    const int hi = (r >> 3) & 1, e = (r & 7) + ((r >> 4) << 3);
    const int lane = nlo + 16 * hi;
    Bp[(((size_t)nt * KSTEPS + kk) * 32 + lane) * 16 + e] = (_Float16)w2[i];
  }
}

// ---------------------------------------------------------------------------
// Kernel 3: LUT GEMM  lut[4112,512] = gelu(g*w1+b1) @ w2 + b2
// One wave per 16x16 D tile. Operands pre-packed: each k-step is two 32B
// lane-contiguous loads (-> 4x global_load_b128) + one v_wmma_f32_16x16x32_f16.
// ---------------------------------------------------------------------------
__global__ void lut_gemm_kernel(const v16h* __restrict__ Ap,
                                const v16h* __restrict__ Bp,
                                const float* __restrict__ b2,
                                float* __restrict__ lut) {
  const int gwave = blockIdx.x * (blockDim.x >> 5) + (threadIdx.x >> 5);
  const int lane  = threadIdx.x & 31;
  const int nt  = gwave & (NTILES - 1);     // 32 N tiles
  const int mt  = gwave / NTILES;           // 257 M tiles
  const int mlo = lane & 15;
  const int hi  = lane >> 4;
  const int ncol = nt * 16 + mlo;           // output column n

  v8f c;
  const float bias = b2[ncol];
#pragma unroll
  for (int v = 0; v < 8; ++v) c[v] = bias;

  const size_t abase = (size_t)mt * KSTEPS * 32 + lane;
  const size_t bbase = (size_t)nt * KSTEPS * 32 + lane;
#pragma unroll
  for (int kk = 0; kk < KSTEPS; ++kk) {
    const v16h a  = Ap[abase + (size_t)kk * 32];
    const v16h bm = Bp[bbase + (size_t)kk * 32];
    c = __builtin_amdgcn_wmma_f32_16x16x32_f16(/*neg_a=*/false, a,
                                               /*neg_b=*/false, bm,
                                               /*c_mod=*/(short)0, c,
                                               /*reuse_a=*/false, /*reuse_b=*/false);
  }
#pragma unroll
  for (int v = 0; v < 8; ++v) {
    const int m = v + 8 * hi;                // D layout: M = v + 8*hi, N = lane&15
    lut[(size_t)(mt * 16 + m) * DHALF + ncol] = c[v];
  }
}

// ---------------------------------------------------------------------------
// Kernel 4: assemble output. One block per token; b128 loads, NT b128 stores
// (output is write-once: bypass near caches, keep emb/lut L2-resident).
// out[token] = [ emb_table[obs] (512) | lut[gap] (512) ]
// ---------------------------------------------------------------------------
__global__ void gather_kernel(const int* __restrict__ obs, const int* __restrict__ gap,
                              const float* __restrict__ emb, const float* __restrict__ lut,
                              float* __restrict__ out) {
  const int token = blockIdx.x;
  const int tid   = threadIdx.x;       // 256 threads -> 256 v4f = 4KB row
  const int o = obs[token];
  const int g = gap[token];
  v4f* dst = (v4f*)(out + (size_t)token * DMODEL);
  if (tid < 128) {                     // wave-uniform split: waves 0-3 vs 4-7
    const v4f* src = (const v4f*)(emb + (size_t)o * DHALF);
    __builtin_nontemporal_store(src[tid], &dst[tid]);
  } else {
    const v4f* src = (const v4f*)(lut + (size_t)g * DHALF);
    __builtin_nontemporal_store(src[tid - 128], &dst[tid]);
  }
}

// ---------------------------------------------------------------------------
// Fallback (only if ws_size is too small for the LUT): direct per-token MLP.
// Deterministic: branch depends only on the constant ws_size.
// ---------------------------------------------------------------------------
__global__ void direct_kernel(const int* __restrict__ obs, const int* __restrict__ gap,
                              const float* __restrict__ emb,
                              const float* __restrict__ w1, const float* __restrict__ b1,
                              const float* __restrict__ w2, const float* __restrict__ b2,
                              float* __restrict__ out) {
  const int token = blockIdx.x;
  const int tid   = threadIdx.x;       // 256
  __shared__ float h[DQ];
  const int g = gap[token];
  const int o = obs[token];
  const float x = (float)g * w1[tid] + b1[tid];
  h[tid] = gelu_exact(x);
  const float2 ce = ((const float2*)(emb + (size_t)o * DHALF))[tid];
  ((float2*)(out + (size_t)token * DMODEL))[tid] = ce;
  __syncthreads();
  float acc0 = b2[2 * tid], acc1 = b2[2 * tid + 1];
  for (int j = 0; j < DQ; ++j) {
    const float hj = h[j];
    acc0 += hj * w2[(size_t)j * DHALF + 2 * tid];
    acc1 += hj * w2[(size_t)j * DHALF + 2 * tid + 1];
  }
  out[(size_t)token * DMODEL + DHALF + 2 * tid]     = acc0;
  out[(size_t)token * DMODEL + DHALF + 2 * tid + 1] = acc1;
}

extern "C" void kernel_launch(void* const* d_in, const int* in_sizes, int n_in,
                              void* d_out, int out_size, void* d_ws, size_t ws_size,
                              hipStream_t stream) {
  (void)in_sizes; (void)n_in; (void)out_size;
  // d_in: 0:x (unused), 1:padding_mask (bool), 2:emb_table, 3:w1, 4:b1, 5:w2, 6:b2
  const uint8_t* mask = (const uint8_t*)d_in[1];   // numpy bool: 1 byte/elt
  const float*   emb  = (const float*)d_in[2];
  const float*   w1   = (const float*)d_in[3];
  const float*   b1   = (const float*)d_in[4];
  const float*   w2   = (const float*)d_in[5];
  const float*   b2   = (const float*)d_in[6];
  float* out = (float*)d_out;

  uint8_t* ws = (uint8_t*)d_ws;
  size_t off = 0;
  int* obs = (int*)(ws + off);            off += (size_t)NTOK * 4;             // 128 KB
  int* gap = (int*)(ws + off);            off += (size_t)NTOK * 4;             // 128 KB
  _Float16* Ap = (_Float16*)(ws + off);   off += (size_t)GROWS * DQ * 2;       // ~2.0 MB (32B-aligned)
  _Float16* Bp = (_Float16*)(ws + off);   off += (size_t)DQ * DHALF * 2;       // 256 KB (32B-aligned)
  float* lut = (float*)(ws + off);        off += (size_t)GROWS * DHALF * 4;    // ~8.4 MB
  const size_t need = off;                                                     // ~10.6 MB

  scan_kernel<<<BB, 256, 0, stream>>>(mask, obs, gap);

  if (ws_size >= need) {
    const int prep_total = GROWS * DQ;                       // 1052672 (> DQ*DHALF)
    prep_kernel<<<(prep_total + 255) / 256, 256, 0, stream>>>(w1, b1, w2, Ap, Bp);
    const int waves = MTILES * NTILES;                       // 257*32 = 8224 tiles
    lut_gemm_kernel<<<waves / 8, 256, 0, stream>>>((const v16h*)Ap, (const v16h*)Bp,
                                                   b2, lut); // 1028 blocks x 8 waves
    gather_kernel<<<NTOK, 256, 0, stream>>>(obs, gap, emb, lut, out);
  } else {
    direct_kernel<<<NTOK, 256, 0, stream>>>(obs, gap, emb, w1, b1, w2, b2, out);
  }
}
